// FreqExtractMoudle_51144470561302
// MI455X (gfx1250) — matrix operand order repaired and checked by reading, not compiled, not run on hardware
//
#include <hip/hip_runtime.h>
#include <hip/hip_bf16.h>

#define BN_EPS 1e-5f
#define NPTS   4096
#define NBATCH 4

typedef __attribute__((ext_vector_type(16))) _Float16 v16h;
typedef __attribute__((ext_vector_type(8)))  float    v8f;

// ---------------- CDNA5 async global->LDS path (guarded) ----------------
#if defined(__AMDGCN__) && __has_builtin(__builtin_amdgcn_global_load_async_to_lds_b32)
#define HAVE_ASYNC_LDS 1
#endif

typedef __attribute__((address_space(1))) int* gint_p;
typedef __attribute__((address_space(3))) int* lint_p;

__device__ __forceinline__ void async_copy_b32(void* lds_dst, const void* g_src) {
#if defined(HAVE_ASYNC_LDS)
  __builtin_amdgcn_global_load_async_to_lds_b32(
      (gint_p)g_src, (lint_p)lds_dst, 0, 0);
#else
  *(float*)lds_dst = *(const float*)g_src;
#endif
}

__device__ __forceinline__ void async_wait_all() {
#if defined(HAVE_ASYNC_LDS)
#if __has_builtin(__builtin_amdgcn_s_wait_asynccnt)
  __builtin_amdgcn_s_wait_asynccnt(0);
#else
  asm volatile("s_wait_asynccnt 0" ::: "memory");
#endif
#endif
}

// ------------- closed-form smallest eigenvector of sym 3x3 -------------
__device__ __forceinline__ void smallest_eigvec(
    float a00, float a01, float a02, float a11, float a12, float a22,
    float& vx, float& vy, float& vz)
{
  float p1 = a01*a01 + a02*a02 + a12*a12;
  float ex, ey, ez;
  if (p1 < 1e-18f) {                       // (near) diagonal
    if (a00 <= a11 && a00 <= a22)      { ex = 1.f; ey = 0.f; ez = 0.f; }
    else if (a11 <= a22)               { ex = 0.f; ey = 1.f; ez = 0.f; }
    else                               { ex = 0.f; ey = 0.f; ez = 1.f; }
  } else {
    float q   = (a00 + a11 + a22) * (1.f/3.f);
    float b00 = a00 - q, b11 = a11 - q, b22 = a22 - q;
    float p2  = b00*b00 + b11*b11 + b22*b22 + 2.f*p1;
    float p   = sqrtf(p2 * (1.f/6.f));
    float ip  = 1.f / p;
    float c00 = b00*ip, c01 = a01*ip, c02 = a02*ip;
    float c11 = b11*ip, c12 = a12*ip, c22 = b22*ip;
    float detB = c00*(c11*c22 - c12*c12) - c01*(c01*c22 - c12*c02)
               + c02*(c01*c12 - c11*c02);
    float r = fminf(1.f, fmaxf(-1.f, 0.5f*detB));
    float phi  = acosf(r) * (1.f/3.f);
    float lmin = q + 2.f*p*cosf(phi + 2.0943951023931953f);  // + 2*pi/3
    float m00 = a00 - lmin, m11 = a11 - lmin, m22 = a22 - lmin;
    float x0 = a01*a12 - a02*m11, y0 = a02*a01 - m00*a12, z0 = m00*m11 - a01*a01;
    float x1 = a01*m22 - a02*a12, y1 = a02*a02 - m00*m22, z1 = m00*a12 - a01*a02;
    float x2 = m11*m22 - a12*a12, y2 = a12*a02 - a01*m22, z2 = a01*a12 - m11*a02;
    float n0 = x0*x0 + y0*y0 + z0*z0;
    float n1 = x1*x1 + y1*y1 + z1*z1;
    float n2 = x2*x2 + y2*y2 + z2*z2;
    if (n0 >= n1 && n0 >= n2) { ex = x0; ey = y0; ez = z0; }
    else if (n1 >= n2)        { ex = x1; ey = y1; ez = z1; }
    else                      { ex = x2; ey = y2; ez = z2; }
    float nn = ex*ex + ey*ey + ez*ez;
    if (nn < 1e-30f) { ex = 1.f; ey = 0.f; ez = 0.f; }
    else { float inv = rsqrtf(nn); ex *= inv; ey *= inv; ez *= inv; }
  }
  float ax = fabsf(ex), ay = fabsf(ey), az = fabsf(ez);
  float pivot = (ax >= ay && ax >= az) ? ex : ((ay >= az) ? ey : ez);
  if (pivot < 0.f) { ex = -ex; ey = -ey; ez = -ez; }
  vx = ex; vy = ey; vz = ez;
}

// ---------------- kernel 1: ball query + cov + eigen + curvature ----------------
__global__ __launch_bounds__(256) void curvature_kernel(
    const float* __restrict__ xyz,       // (B, N, 3)
    const float* __restrict__ psmall,
    const float* __restrict__ plarge,
    float* __restrict__ curv)            // (B, N)
{
  __shared__ float4 pts[NPTS];           // {x,y,z,|p|^2}  == 64 KB LDS
  const int b  = blockIdx.y;
  const int qi = blockIdx.x * blockDim.x + threadIdx.x;
  const float* xb = xyz + (size_t)b * NPTS * 3;

  // Stage the whole batch point cloud into LDS via CDNA5 async copies.
  for (int j = threadIdx.x; j < NPTS; j += blockDim.x) {
    async_copy_b32(&pts[j].x, xb + 3*j + 0);
    async_copy_b32(&pts[j].y, xb + 3*j + 1);
    async_copy_b32(&pts[j].z, xb + 3*j + 2);
  }
  async_wait_all();
  __syncthreads();
  for (int j = threadIdx.x; j < NPTS; j += blockDim.x) {
    float4 p = pts[j];
    pts[j].w = p.x*p.x + p.y*p.y + p.z*p.z;
  }
  __syncthreads();

  const float r2s = psmall[0]*psmall[0];
  const float r2l = plarge[0]*plarge[0];
  const float4 q = pts[qi];

  int cs = 0, cl = 0;
  float fsx=0,fsy=0,fsz=0, flx=0,fly=0,flz=0;
  float ssx=0,ssy=0,ssz=0, sxx=0,sxy=0,sxz=0,syy=0,syz=0,szz=0;
  float tsx=0,tsy=0,tsz=0, txx=0,txy=0,txz=0,tyy=0,tyz=0,tzz=0;

  for (int j = 0; j < NPTS; ++j) {
    float4 p = pts[j];                                  // wave-broadcast b128
    float dot  = q.x*p.x + q.y*p.y + q.z*p.z;
    float dist = q.w + p.w - 2.f*dot;                   // matches reference formula
    if (dist <= r2l) {
      if (cl == 0) { flx = p.x; fly = p.y; flz = p.z; }
      if (cl < 64) {
        tsx += p.x; tsy += p.y; tsz += p.z;
        txx += p.x*p.x; txy += p.x*p.y; txz += p.x*p.z;
        tyy += p.y*p.y; tyz += p.y*p.z; tzz += p.z*p.z;
        ++cl;
      }
      if (dist <= r2s) {
        if (cs == 0) { fsx = p.x; fsy = p.y; fsz = p.z; }
        if (cs < 32) {
          ssx += p.x; ssy += p.y; ssz += p.z;
          sxx += p.x*p.x; sxy += p.x*p.y; sxz += p.x*p.z;
          syy += p.y*p.y; syz += p.y*p.z; szz += p.z*p.z;
          ++cs;
        }
      }
      if (cs >= 32 && cl >= 64) break;
    }
  }
  // pad with first in-radius point (reference pads with group_idx[...,0])
  if (cs < 32) {
    float n = (float)(32 - cs);
    ssx += n*fsx; ssy += n*fsy; ssz += n*fsz;
    sxx += n*fsx*fsx; sxy += n*fsx*fsy; sxz += n*fsx*fsz;
    syy += n*fsy*fsy; syz += n*fsy*fsz; szz += n*fsz*fsz;
  }
  if (cl < 64) {
    float n = (float)(64 - cl);
    tsx += n*flx; tsy += n*fly; tsz += n*flz;
    txx += n*flx*flx; txy += n*flx*fly; txz += n*flx*flz;
    tyy += n*fly*fly; tyz += n*fly*flz; tzz += n*flz*flz;
  }

  const float is = 1.f/32.f, il = 1.f/64.f;
  float nsx, nsy, nsz, nlx, nly, nlz;
  smallest_eigvec(sxx - ssx*ssx*is, sxy - ssx*ssy*is, sxz - ssx*ssz*is,
                  syy - ssy*ssy*is, syz - ssy*ssz*is, szz - ssz*ssz*is,
                  nsx, nsy, nsz);
  smallest_eigvec(txx - tsx*tsx*il, txy - tsx*tsy*il, txz - tsx*tsz*il,
                  tyy - tsy*tsy*il, tyz - tsy*tsz*il, tzz - tsz*tsz*il,
                  nlx, nly, nlz);

  float dx = 0.5f*(nsx - nlx), dy = 0.5f*(nsy - nly), dz = 0.5f*(nsz - nlz);
  curv[(size_t)b*NPTS + qi] = sqrtf(dx*dx + dy*dy + dz*dz);
}

// ---------------- kernel 2: pointwise MLP via WMMA (one wave / 16 points) ----------------
// All WMMA operands are packed from zero-padded f16 LDS tiles -> unconditional
// ds loads, no lane-divergent global traffic.
__global__ __launch_bounds__(32) void mlp_wmma_kernel(
    const float* __restrict__ curv,
    const float* __restrict__ w1, const float* __restrict__ b1,
    const float* __restrict__ g1, const float* __restrict__ be1,
    const float* __restrict__ m1, const float* __restrict__ v1,
    const float* __restrict__ w2, const float* __restrict__ b2,
    const float* __restrict__ g2, const float* __restrict__ be2,
    const float* __restrict__ m2, const float* __restrict__ v2,
    const float* __restrict__ w3, const float* __restrict__ b3,
    const float* __restrict__ g3, const float* __restrict__ be3,
    const float* __restrict__ m3, const float* __restrict__ v3,
    const float* __restrict__ w4, const float* __restrict__ b4,
    const float* __restrict__ g4, const float* __restrict__ be4,
    const float* __restrict__ m4, const float* __restrict__ v4,
    float* __restrict__ out)
{
  const int lane = threadIdx.x;
  const int base = blockIdx.x * 16;          // 16 points per wave
  __shared__ _Float16 h1h[16][32];           // layer-1 output, K padded to 32
  __shared__ _Float16 h2h[16][32];           // layer-2 output (K=32 exact)
  __shared__ float    h3[16][16];
  __shared__ _Float16 w2h[32][32];           // w2^T tile, K padded 16->32
  __shared__ _Float16 w3h[16][32];           // w3^T tile, K=32 exact

  // ---- stage weights into zero-padded f16 LDS tiles (all 32 lanes) ----
  #pragma unroll
  for (int i = lane; i < 32*16; i += 32) {   // w2 is (32 out, 16 in)
    int oc = i >> 4, k = i & 15;
    w2h[oc][k]      = (_Float16)w2[i];
    w2h[oc][k + 16] = (_Float16)0.0f;
  }
  #pragma unroll
  for (int i = lane; i < 16*32; i += 32) {   // w3 is (16 out, 32 in)
    w3h[i >> 5][i & 31] = (_Float16)w3[i];
  }

  // ---- layer 1 (1 -> 16), scalar: lane p handles point p ----
  if (lane < 16) {
    float c = curv[base + lane];
    #pragma unroll
    for (int k = 0; k < 16; ++k) {
      float s = g1[k] * rsqrtf(v1[k] + BN_EPS);
      float h = (c * w1[k] + b1[k] - m1[k]) * s + be1[k];
      h1h[lane][k]      = (_Float16)fmaxf(h, 0.f);
      h1h[lane][k + 16] = (_Float16)0.0f;
    }
  }
  __syncthreads();

  const int M  = lane & 15;
  const int hi = (lane >> 4) & 1;

  // ---- layer 2: A = h1 (16x16 zero-padded to K=32), B = w2^T, two N tiles ----
  v16h a;
  #pragma unroll
  for (int e = 0; e < 16; ++e) {             // ISA 16-bit A 16x32 layout
    int gI = e >> 1;
    int K  = ((gI & 3) << 1) + (e & 1) + hi*8 + (gI >> 2)*16;
    a[e] = h1h[M][K];
  }
  #pragma unroll
  for (int t = 0; t < 2; ++t) {
    int oc = M + 16*t;
    v16h bm;
    #pragma unroll
    for (int e = 0; e < 16; ++e) {           // ISA 16-bit B 32x16 layout
      bm[e] = w2h[oc][(e & 7) + hi*8 + (e >> 3)*16];
    }
    v8f d = {};
    d = __builtin_amdgcn_wmma_f32_16x16x32_f16(false, a, false, bm,
                                               (short)0, d, false, false);
    float s  = g2[oc] * rsqrtf(v2[oc] + BN_EPS);
    float bb = (b2[oc] - m2[oc]) * s + be2[oc];
    #pragma unroll
    for (int r = 0; r < 8; ++r) {            // ISA f32 C/D 16x16 layout
      int Mr = r + hi*8;
      h2h[Mr][oc] = (_Float16)fmaxf(d[r]*s + bb, 0.f);
    }
  }
  __syncthreads();

  // ---- layer 3: A = h2 (16x32, K=32 exact), B = w3^T (32->16) ----
  v16h a3;
  #pragma unroll
  for (int e = 0; e < 16; ++e) {
    int gI = e >> 1;
    int K  = ((gI & 3) << 1) + (e & 1) + hi*8 + (gI >> 2)*16;
    a3[e] = h2h[M][K];
  }
  {
    int oc = M;
    v16h bm;
    #pragma unroll
    for (int e = 0; e < 16; ++e) {
      bm[e] = w3h[oc][(e & 7) + hi*8 + (e >> 3)*16];
    }
    v8f d = {};
    d = __builtin_amdgcn_wmma_f32_16x16x32_f16(false, a3, false, bm,
                                               (short)0, d, false, false);
    float s  = g3[oc] * rsqrtf(v3[oc] + BN_EPS);
    float bb = (b3[oc] - m3[oc]) * s + be3[oc];
    #pragma unroll
    for (int r = 0; r < 8; ++r) {
      int Mr = r + hi*8;
      h3[Mr][oc] = fmaxf(d[r]*s + bb, 0.f);
    }
  }
  __syncthreads();

  // ---- layer 4: 16 -> 1, no relu ----
  if (lane < 16) {
    float acc = 0.f;
    #pragma unroll
    for (int k = 0; k < 16; ++k) acc += h3[lane][k] * w4[k];
    float s = g4[0] * rsqrtf(v4[0] + BN_EPS);
    out[base + lane] = (acc + b4[0] - m4[0]) * s + be4[0];
  }
}

extern "C" void kernel_launch(void* const* d_in, const int* in_sizes, int n_in,
                              void* d_out, int out_size, void* d_ws, size_t ws_size,
                              hipStream_t stream) {
  (void)in_sizes; (void)n_in; (void)out_size; (void)ws_size;
  const float* x      = (const float*)d_in[0];
  const float* smallr = (const float*)d_in[1];
  const float* larger = (const float*)d_in[2];
  float* curv = (float*)d_ws;                  // B*N floats scratch

  dim3 grid1(NPTS / 256, NBATCH);
  curvature_kernel<<<grid1, 256, 0, stream>>>(x, smallr, larger, curv);

  mlp_wmma_kernel<<<(NBATCH * NPTS) / 16, 32, 0, stream>>>(
      curv,
      (const float*)d_in[3],  (const float*)d_in[4],  (const float*)d_in[5],
      (const float*)d_in[6],  (const float*)d_in[7],  (const float*)d_in[8],
      (const float*)d_in[9],  (const float*)d_in[10], (const float*)d_in[11],
      (const float*)d_in[12], (const float*)d_in[13], (const float*)d_in[14],
      (const float*)d_in[15], (const float*)d_in[16], (const float*)d_in[17],
      (const float*)d_in[18], (const float*)d_in[19], (const float*)d_in[20],
      (const float*)d_in[21], (const float*)d_in[22], (const float*)d_in[23],
      (const float*)d_in[24], (const float*)d_in[25], (const float*)d_in[26],
      (float*)d_out);
}
